// Model_678604832982
// MI455X (gfx1250) — matrix operand order repaired and checked by reading, not compiled, not run on hardware
//
#include <hip/hip_runtime.h>
#include <hip/hip_bf16.h>

// Fused football-attention pipeline for gfx1250 (MI455X).
// All GEMMs via v_wmma_f32_16x16x32_f16 (f16 operands, f32 accumulate);
// LayerNorm / softmax / attention-weighted sums in f32 VALU.
// One wave (32 threads) per 16-batch tile; everything staged through LDS;
// zero global workspace.

#define NPLY 11
#define F_IN 36
#define DIM  48
#define KP0  64    // padded K for the 36- and 48-deep GEMMs
#define DIM2 96
#define MT   16    // batch rows per wave tile

typedef __attribute__((ext_vector_type(16))) _Float16 v16h;
typedef __attribute__((ext_vector_type(8)))  _Float16 v8h;
typedef __attribute__((ext_vector_type(8)))  float    v8f;

union FragU { v16h v; v8h h[2]; };

// A fragment (16xK tile, row-major [16][KP] f16 in LDS).
// ISA layout: lane L holds M=L%16; half j -> k = kbase + 8*(L>=16) + (j%8) + 16*(j>=8)
__device__ __forceinline__ v16h frag_A(const _Float16* a, int KP, int lane, int kbase) {
  FragU f;
  const _Float16* p = a + (lane & 15) * KP + kbase + ((lane >> 4) << 3);
  f.h[0] = *(const v8h*)(p);
  f.h[1] = *(const v8h*)(p + 16);
  return f.v;
}

// B fragment (Kx16 tile, stored N-major [16 N][KP] f16 in LDS, i.e. W transposed).
// ISA layout: lane L holds N=L%16; half j -> k = kbase + 8*(L>=16) + (j%8) + 16*(j>=8)
__device__ __forceinline__ v16h frag_B(const _Float16* b, int KP, int lane, int kbase) {
  FragU f;
  const _Float16* p = b + (lane & 15) * KP + kbase + ((lane >> 4) << 3);
  f.h[0] = *(const v8h*)(p);
  f.h[1] = *(const v8h*)(p + 16);
  return f.v;
}

// C = A[16][KP] @ B^T (B stored [NOUT][KP]) -> C[16][NOUT] (f32 or f16 dest).
// C layout: lane L -> N=L%16, VGPR r -> M = r + 8*(L>=16)
template<int KP, int NOUT, typename T>
__device__ __forceinline__ void gemm16(const _Float16* A, const _Float16* B, T* C, int lane) {
  #pragma unroll
  for (int nt = 0; nt < NOUT / 16; ++nt) {
    v8f c = {0.f, 0.f, 0.f, 0.f, 0.f, 0.f, 0.f, 0.f};
    #pragma unroll
    for (int ks = 0; ks < KP / 32; ++ks) {
      v16h a = frag_A(A, KP, lane, ks * 32);
      v16h b = frag_B(B + nt * 16 * KP, KP, lane, ks * 32);
      c = __builtin_amdgcn_wmma_f32_16x16x32_f16(false, a, false, b, (short)0, c, false, false);
    }
    const int n  = nt * 16 + (lane & 15);
    const int mb = (lane >> 4) << 3;
    #pragma unroll
    for (int r = 0; r < 8; ++r) C[(mb + r) * NOUT + n] = (T)c[r];
  }
}

// Stage a [16][36] f32 global tile into [16][64] f16 LDS (zero K-pad).
__device__ __forceinline__ void load_tile36(const float* __restrict__ g, int rowStride,
                                            _Float16* dst, int lane) {
  for (int idx = lane; idx < MT * KP0; idx += 32) {
    const int m = idx >> 6, k = idx & 63;
    dst[idx] = (_Float16)((k < F_IN) ? g[m * rowStride + k] : 0.f);
  }
}

// W global [IN][OUT] f32 -> LDS [OUT][KP] f16 (transposed, zero K-pad).
__device__ __forceinline__ void load_Wt(const float* __restrict__ W, int IN, int OUT, int KP,
                                        _Float16* lds, int t) {
  for (int idx = t; idx < OUT * KP; idx += 32) {
    const int o = idx / KP, k = idx - o * KP;
    lds[idx] = (_Float16)((k < IN) ? W[k * OUT + o] : 0.f);
  }
}

// y = relu(LN(C+bias)) ; C f32 [16][48] -> E f16 [16][64] (zero pad cols 48..63)
__device__ __forceinline__ void ln_relu(const float* C, const float* bias, const float* g,
                                        const float* bet, _Float16* E, int lane) {
  if (lane < MT) {
    const int m = lane;
    float s = 0.f;
    for (int k = 0; k < DIM; ++k) s += C[m * DIM + k] + bias[k];
    const float mu = s * (1.f / DIM);
    float v = 0.f;
    for (int k = 0; k < DIM; ++k) { float d = C[m * DIM + k] + bias[k] - mu; v += d * d; }
    const float rs = rsqrtf(v * (1.f / DIM) + 1e-5f);
    for (int k = 0; k < DIM; ++k) {
      float y = (C[m * DIM + k] + bias[k] - mu) * rs * g[k] + bet[k];
      E[m * KP0 + k] = (_Float16)fmaxf(y, 0.f);
    }
  } else {
    const int m = lane - 16;
    #pragma unroll
    for (int k = DIM; k < KP0; ++k) E[m * KP0 + k] = (_Float16)0.f;
  }
}

__device__ __forceinline__ void softmax_rows(float* a, int lane) {
  if (lane < MT) {
    float* r = a + lane * NPLY;
    float mx = r[0];
    #pragma unroll
    for (int i = 1; i < NPLY; ++i) mx = fmaxf(mx, r[i]);
    float s = 0.f;
    #pragma unroll
    for (int i = 0; i < NPLY; ++i) { float e = __expf(r[i] - mx); r[i] = e; s += e; }
    const float inv = 1.f / s;
    #pragma unroll
    for (int i = 0; i < NPLY; ++i) r[i] *= inv;
  }
}

__global__ __launch_bounds__(32) void fused_attention(
    const float* __restrict__ opp, const float* __restrict__ left, const float* __restrict__ right,
    const float* __restrict__ Wopp, const float* __restrict__ bopp,
    const float* __restrict__ Wleft, const float* __restrict__ bleft,
    const float* __restrict__ Wright, const float* __restrict__ bright,
    const float* __restrict__ lng, const float* __restrict__ lnb,
    const float* __restrict__ Wq1, const float* __restrict__ Wk1, const float* __restrict__ Wv1,
    const float* __restrict__ Wq2, const float* __restrict__ Wk2,
    float* __restrict__ out1, float* __restrict__ out2) {
  // ---- LDS: weights (~72KB) + tile scratch (~54KB) = ~126KB ----
  __shared__ __align__(16) _Float16 sWopp[DIM * KP0], sWleft[DIM * KP0], sWright[DIM * KP0];
  __shared__ __align__(16) _Float16 sWq1[DIM * KP0], sWk1[DIM * KP0], sWv1[DIM * KP0];
  __shared__ __align__(16) _Float16 sWq2[DIM2 * DIM2], sWk2[DIM2 * DIM2];
  __shared__ float sb[3][DIM], sg[DIM], sbeta[DIM];
  __shared__ __align__(16) _Float16 Abuf[MT * KP0];     // staged input tile
  __shared__ __align__(16) _Float16 Ebuf[MT * KP0];     // relu(LN(..)) embed, K-padded
  __shared__ __align__(16) _Float16 emb2[MT * DIM2];    // concat[v1, att@left_v1]
  __shared__ __align__(16) _Float16 q1h[MT * DIM], v1h[MT * DIM], q2h[MT * DIM2];
  __shared__ __align__(16) _Float16 lK1[NPLY][MT * DIM], lV1[NPLY][MT * DIM];
  __shared__ __align__(16) float Cbuf[MT * DIM2];       // f32 GEMM output
  __shared__ float att[MT * NPLY], lg2[MT * NPLY];

  const int lane = threadIdx.x & 31;
  const int base = blockIdx.x * MT;
  const int rstr = NPLY * F_IN;  // 396

  load_Wt(Wopp,  F_IN, DIM, KP0, sWopp,  threadIdx.x);
  load_Wt(Wleft, F_IN, DIM, KP0, sWleft, threadIdx.x);
  load_Wt(Wright,F_IN, DIM, KP0, sWright,threadIdx.x);
  load_Wt(Wq1, DIM, DIM, KP0, sWq1, threadIdx.x);
  load_Wt(Wk1, DIM, DIM, KP0, sWk1, threadIdx.x);
  load_Wt(Wv1, DIM, DIM, KP0, sWv1, threadIdx.x);
  load_Wt(Wq2, DIM2, DIM2, DIM2, sWq2, threadIdx.x);
  load_Wt(Wk2, DIM2, DIM2, DIM2, sWk2, threadIdx.x);
  for (int k = lane; k < DIM; k += 32) {
    sb[0][k] = bopp[k]; sb[1][k] = bleft[k]; sb[2][k] = bright[k];
    sg[k] = lng[k]; sbeta[k] = lnb[k];
  }
  __syncthreads();

  // ---- left team: e -> k1, v1 (kept resident for cross-attention) ----
  for (int p = 0; p < NPLY; ++p) {
    if (p + 1 < NPLY && lane < MT)
      __builtin_prefetch(left + (size_t)(base + lane) * rstr + (p + 1) * F_IN, 0, 1);
    load_tile36(left + (size_t)base * rstr + p * F_IN, rstr, Abuf, lane);
    __syncthreads();
    gemm16<KP0, DIM, float>(Abuf, sWleft, Cbuf, lane);
    __syncthreads();
    ln_relu(Cbuf, sb[1], sg, sbeta, Ebuf, lane);
    __syncthreads();
    gemm16<KP0, DIM, _Float16>(Ebuf, sWk1, lK1[p], lane);
    gemm16<KP0, DIM, _Float16>(Ebuf, sWv1, lV1[p], lane);
    __syncthreads();
  }

  // ---- opp: e -> q1, v1 -> att over left -> emb2 -> q2 ----
  load_tile36(opp + (size_t)base * F_IN, F_IN, Abuf, lane);
  __syncthreads();
  gemm16<KP0, DIM, float>(Abuf, sWopp, Cbuf, lane);
  __syncthreads();
  ln_relu(Cbuf, sb[0], sg, sbeta, Ebuf, lane);
  __syncthreads();
  gemm16<KP0, DIM, _Float16>(Ebuf, sWq1, q1h, lane);
  gemm16<KP0, DIM, _Float16>(Ebuf, sWv1, v1h, lane);
  __syncthreads();
  for (int idx = lane; idx < MT * NPLY; idx += 32) {  // logits q1.k1
    const int m = idx / NPLY, p = idx - m * NPLY;
    const _Float16* q = q1h + m * DIM;
    const _Float16* kk = lK1[p] + m * DIM;
    float s = 0.f;
    for (int d = 0; d < DIM; ++d) s += (float)q[d] * (float)kk[d];
    att[idx] = s;
  }
  __syncthreads();
  softmax_rows(att, lane);
  __syncthreads();
  for (int idx = lane; idx < MT * DIM2; idx += 32) {  // emb2 = [v1 | att@left_v1]
    const int m = idx / DIM2, d = idx - m * DIM2;
    float v;
    if (d < DIM) v = (float)v1h[m * DIM + d];
    else {
      const int dd = d - DIM; v = 0.f;
      for (int p = 0; p < NPLY; ++p) v += att[m * NPLY + p] * (float)lV1[p][m * DIM + dd];
    }
    emb2[idx] = (_Float16)v;
  }
  __syncthreads();
  gemm16<DIM2, DIM2, _Float16>(emb2, sWq2, q2h, lane);
  __syncthreads();

  // ---- right team: per player q1,v1 -> att_to_left (output) -> emb2 -> k2 -> logits2 ----
  for (int p = 0; p < NPLY; ++p) {
    if (p + 1 < NPLY && lane < MT)
      __builtin_prefetch(right + (size_t)(base + lane) * rstr + (p + 1) * F_IN, 0, 1);
    load_tile36(right + (size_t)base * rstr + p * F_IN, rstr, Abuf, lane);
    __syncthreads();
    gemm16<KP0, DIM, float>(Abuf, sWright, Cbuf, lane);
    __syncthreads();
    ln_relu(Cbuf, sb[2], sg, sbeta, Ebuf, lane);
    __syncthreads();
    gemm16<KP0, DIM, _Float16>(Ebuf, sWq1, q1h, lane);
    gemm16<KP0, DIM, _Float16>(Ebuf, sWv1, v1h, lane);
    __syncthreads();
    for (int idx = lane; idx < MT * NPLY; idx += 32) {
      const int m = idx / NPLY, pl = idx - m * NPLY;
      const _Float16* q = q1h + m * DIM;
      const _Float16* kk = lK1[pl] + m * DIM;
      float s = 0.f;
      for (int d = 0; d < DIM; ++d) s += (float)q[d] * (float)kk[d];
      att[idx] = s;
    }
    __syncthreads();
    softmax_rows(att, lane);
    __syncthreads();
    for (int idx = lane; idx < MT * NPLY; idx += 32) {  // right_att_to_left output
      const int m = idx / NPLY, pl = idx - m * NPLY;
      out2[((size_t)(base + m) * NPLY + p) * NPLY + pl] = att[idx];
    }
    for (int idx = lane; idx < MT * DIM2; idx += 32) {  // right_emb2
      const int m = idx / DIM2, d = idx - m * DIM2;
      float v;
      if (d < DIM) v = (float)v1h[m * DIM + d];
      else {
        const int dd = d - DIM; v = 0.f;
        for (int pl = 0; pl < NPLY; ++pl) v += att[m * NPLY + pl] * (float)lV1[pl][m * DIM + dd];
      }
      emb2[idx] = (_Float16)v;
    }
    __syncthreads();
    gemm16<DIM2, DIM2, float>(emb2, sWk2, Cbuf, lane);  // k2 for this player
    __syncthreads();
    if (lane < MT) {  // logits2[:,p] = q2 . k2
      const _Float16* q = q2h + lane * DIM2;
      const float* kk = Cbuf + lane * DIM2;
      float s = 0.f;
      for (int d = 0; d < DIM2; ++d) s += (float)q[d] * kk[d];
      lg2[lane * NPLY + p] = s;
    }
    __syncthreads();
  }

  softmax_rows(lg2, lane);
  if (lane < MT) {
    const float* r = lg2 + lane * NPLY;
    for (int i = 0; i < NPLY; ++i) out1[(size_t)(base + lane) * NPLY + i] = r[i];
  }
}

extern "C" void kernel_launch(void* const* d_in, const int* in_sizes, int n_in,
                              void* d_out, int out_size, void* d_ws, size_t ws_size,
                              hipStream_t stream) {
  (void)n_in; (void)out_size; (void)d_ws; (void)ws_size;
  const float* opp    = (const float*)d_in[0];
  const float* left   = (const float*)d_in[1];
  const float* right  = (const float*)d_in[2];
  const float* Wopp   = (const float*)d_in[3];
  const float* bopp   = (const float*)d_in[4];
  const float* Wleft  = (const float*)d_in[5];
  const float* bleft  = (const float*)d_in[6];
  const float* Wright = (const float*)d_in[7];
  const float* bright = (const float*)d_in[8];
  const float* lng    = (const float*)d_in[9];
  const float* lnb    = (const float*)d_in[10];
  const float* Wq1    = (const float*)d_in[11];
  const float* Wk1    = (const float*)d_in[12];
  const float* Wv1    = (const float*)d_in[13];
  const float* Wq2    = (const float*)d_in[14];
  const float* Wk2    = (const float*)d_in[15];

  const int Bv = in_sizes[0] / F_IN;        // 65536
  float* out1 = (float*)d_out;              // opp_att_to_right  [B,11]
  float* out2 = out1 + (size_t)Bv * NPLY;   // right_att_to_left [B,11,11]

  dim3 grid(Bv / MT), block(32);
  fused_attention<<<grid, block, 0, stream>>>(opp, left, right, Wopp, bopp, Wleft, bleft,
                                              Wright, bright, lng, lnb, Wq1, Wk1, Wv1,
                                              Wq2, Wk2, out1, out2);
}
